// ChebyshevApproximation_15633680958241
// MI455X (gfx1250) — matrix operand order repaired and checked by reading, not compiled, not run on hardware
//
#include <hip/hip_runtime.h>

#define DEG   1024
#define NOBS  65536
#define MP1   2049
#define PI_D  3.14159265358979323846
#define PI_F  3.14159265358979323846f

typedef __bf16 v16bf __attribute__((ext_vector_type(16)));
typedef float  v8f   __attribute__((ext_vector_type(8)));

// Workspace layout (bytes):
//   [0,     4096)  w0   float[1024]
//   [4096,  8192)  w1   float[1024]
//   [8192, 12288)  idx  int[1024]
//   [16384, 16384+2MB) basis bf16 in B-fragment order:
//       frag(s=kstep 0..31, ntile 0..63) -> 32 lanes x 16 halfs (32B/lane)
//       lane L, half q: K = s*32 + 8*(L>=16) + (q&7) + 16*(q>=8) ; N = ntile*16 + (L&15)

__global__ void cheb_nodes_kernel(const float* __restrict__ x,
                                  float* __restrict__ w0,
                                  float* __restrict__ w1,
                                  int* __restrict__ idxv) {
    int j = blockIdx.x * blockDim.x + threadIdx.x;
    if (j >= DEG) return;
    // ascending-sorted Chebyshev node: -cos(pi*(j+0.5)/DEG)
    float node = -__builtin_cosf(PI_F * (j + 0.5f) * (1.0f / DEG));
    // searchsorted(x, node, side='right') - 1
    int lo = 0, hi = MP1;
    while (lo < hi) {
        int mid = (lo + hi) >> 1;
        if (x[mid] <= node) lo = mid + 1; else hi = mid;
    }
    int i0 = lo - 1;
    if (i0 < 0) i0 = 0;
    if (i0 > MP1 - 2) i0 = MP1 - 2;
    float a = x[i0], b = x[i0 + 1];
    float t = (node - a) / (b - a);
    w0[j] = 1.0f - t;
    w1[j] = t;
    idxv[j] = i0;
}

__global__ void cheb_basis_kernel(__bf16* __restrict__ B) {
    // p = half-element index into fragment-ordered basis buffer
    int p = blockIdx.x * blockDim.x + threadIdx.x;     // 0 .. 1048575
    int q    = p & 15;
    int lane = (p >> 4) & 31;
    int ft   = p >> 9;                                 // frag = s*64 + ntile
    int ntile = ft & 63;
    int s     = ft >> 6;
    int k = s * 32 + ((lane >> 4) << 3) + (q & 7) + ((q >> 3) << 4);  // K index (node j)
    int n = ntile * 16 + (lane & 15);                  // N index (coeff)
    // acos(node_k) = pi - pi*(k+0.5)/DEG  (nodes sorted ascending)
    double theta = PI_D - PI_D * ((double)k + 0.5) / (double)DEG;
    double norm  = ((n == 0) ? 1.0 : 2.0) / (double)DEG;
    float val = (float)(norm * cos((double)n * theta));
    B[p] = (__bf16)val;
}

#define WMMA_BF16(A_, B_, C_) \
    __builtin_amdgcn_wmma_f32_16x16x32_bf16(false, (A_), false, (B_), (short)0, (C_), false, false)

__global__ void __launch_bounds__(256, 4)
cheb_gemm_kernel(const float* __restrict__ y,
                 const float* __restrict__ w0,
                 const float* __restrict__ w1,
                 const int* __restrict__ idxv,
                 const __bf16* __restrict__ B,
                 float* __restrict__ out) {
    extern __shared__ char smem[];                // 64 rows x 1024 K bf16 = 128 KB
    const int tid   = threadIdx.x;
    const int wgRow = blockIdx.x * 64;

    // ---- Phase 1: gather + lerp y at nodes, scatter to LDS in A-frag layout
    for (int e = tid; e < 64 * DEG; e += 256) {
        int row = e >> 10;                        // 0..63
        int j   = e & (DEG - 1);                  // node index (K) — coalesced
        int i0  = idxv[j];
        const float* yr = y + (size_t)(wgRow + row) * MP1 + i0;
        float v = fmaf(w1[j], yr[1], w0[j] * yr[0]);
        // A-fragment slot (16-bit A 16x32, ISA 7.12.2)
        int g = row >> 4, r = row & 15;
        int s = j >> 5,  kk = j & 31;
        int quarter = kk >> 3;
        int lane = r + ((quarter & 1) << 4);
        int vq   = ((quarter >> 1) << 2) + ((kk & 7) >> 1);
        int h    = kk & 1;
        int byteoff = ((g * 32 + s) << 10) + (lane << 5) + (vq << 2) + (h << 1);
        *(__bf16*)(smem + byteoff) = (__bf16)v;
    }
    __syncthreads();

    // ---- Phase 2: 8 waves; wave -> (row-group g = wave&3, N-half = wave>>2)
    const int wave = tid >> 5, lane = tid & 31;
    const int g     = wave & 3;
    const int nhalf = wave >> 2;
    const char* Ab = smem + (size_t)(g * 32) * 1024 + lane * 32;   // +s*1024
    const char* Bb = (const char*)B + lane * 32;

    #pragma unroll 1
    for (int ngrp = 0; ngrp < 8; ++ngrp) {
        int nt0 = nhalf * 32 + ngrp * 4;          // 4 consecutive 16-col tiles
        const char* bs0 = Bb + (size_t)nt0 * 1024;   // + s*65536 per K-step

        v16bf Abuf[4];
        v16bf Bbuf[4][4];
        // prologue: load K-steps 0..3 into the 4 pipeline stages
        #pragma unroll
        for (int k = 0; k < 4; ++k) {
            Abuf[k] = *(const v16bf*)(Ab + k * 1024);
            const char* bsk = bs0 + (size_t)k * 65536;
            Bbuf[k][0] = *(const v16bf*)(bsk);
            Bbuf[k][1] = *(const v16bf*)(bsk + 1024);
            Bbuf[k][2] = *(const v16bf*)(bsk + 2048);
            Bbuf[k][3] = *(const v16bf*)(bsk + 3072);
        }

        v8f acc0 = {}, acc1 = {}, acc2 = {}, acc3 = {};

        // steady state: consume stage k (step s+k), refill with step s+k+4.
        // s = 0,4,...,24 -> consumes steps 0..27, prefetches up to step 31.
        #pragma unroll 1
        for (int s = 0; s < 28; s += 4) {
            #pragma unroll
            for (int k = 0; k < 4; ++k) {
                acc0 = WMMA_BF16(Abuf[k], Bbuf[k][0], acc0);
                acc1 = WMMA_BF16(Abuf[k], Bbuf[k][1], acc1);
                acc2 = WMMA_BF16(Abuf[k], Bbuf[k][2], acc2);
                acc3 = WMMA_BF16(Abuf[k], Bbuf[k][3], acc3);
                int sn = s + 4 + k;
                Abuf[k] = *(const v16bf*)(Ab + sn * 1024);
                const char* bsn = bs0 + (size_t)sn * 65536;
                Bbuf[k][0] = *(const v16bf*)(bsn);
                Bbuf[k][1] = *(const v16bf*)(bsn + 1024);
                Bbuf[k][2] = *(const v16bf*)(bsn + 2048);
                Bbuf[k][3] = *(const v16bf*)(bsn + 3072);
            }
        }
        // epilogue: steps 28..31, no more loads
        #pragma unroll
        for (int k = 0; k < 4; ++k) {
            acc0 = WMMA_BF16(Abuf[k], Bbuf[k][0], acc0);
            acc1 = WMMA_BF16(Abuf[k], Bbuf[k][1], acc1);
            acc2 = WMMA_BF16(Abuf[k], Bbuf[k][2], acc2);
            acc3 = WMMA_BF16(Abuf[k], Bbuf[k][3], acc3);
        }

        // D layout: lanes 0-15 -> M=d, lanes 16-31 -> M=8+d; N = lane&15
        int m0   = (lane >> 4) << 3;
        int col0 = nt0 * 16 + (lane & 15);
        int rowb = wgRow + g * 16 + m0;
        #pragma unroll
        for (int d = 0; d < 8; ++d) {
            float* op = out + (size_t)(rowb + d) * DEG + col0;
            __builtin_nontemporal_store(acc0[d], op);
            __builtin_nontemporal_store(acc1[d], op + 16);
            __builtin_nontemporal_store(acc2[d], op + 32);
            __builtin_nontemporal_store(acc3[d], op + 48);
        }
    }
}

extern "C" void kernel_launch(void* const* d_in, const int* in_sizes, int n_in,
                              void* d_out, int out_size, void* d_ws, size_t ws_size,
                              hipStream_t stream) {
    const float* x = (const float*)d_in[0];   // [2049]
    const float* y = (const float*)d_in[1];   // [65536 x 2049]
    float* out = (float*)d_out;               // [65536 x 1024]
    char* ws = (char*)d_ws;
    float* w0   = (float*)(ws + 0);
    float* w1   = (float*)(ws + 4096);
    int*   idxv = (int*)(ws + 8192);
    __bf16* B   = (__bf16*)(ws + 16384);      // 2 MB fragment-ordered basis

    cheb_nodes_kernel<<<4, 256, 0, stream>>>(x, w0, w1, idxv);
    cheb_basis_kernel<<<(DEG * DEG) / 256, 256, 0, stream>>>(B);
    cheb_gemm_kernel<<<NOBS / 64, 256, 64 * DEG * 2, stream>>>(y, w0, w1, idxv, B, out);
}
